// LocalDynamics_29935922053643
// MI455X (gfx1250) — matrix operand order repaired and checked by reading, not compiled, not run on hardware
//
#include <hip/hip_runtime.h>
#include <math.h>

#define N_OBJ   100000
#define E_EDGES 800000
#define D_DIM   64
#define NTILES  (E_EDGES / 16)          // 50,000 16-edge tiles

typedef __attribute__((ext_vector_type(16))) _Float16 v16h;
typedef __attribute__((ext_vector_type(8)))  _Float16 v8h;
typedef __attribute__((ext_vector_type(8)))  float    v8f;

// ---- workspace layout ----
#define DELTA_BYTES (N_OBJ * D_DIM * 4)     // 25,600,000 (256-aligned)
#define W16_HALFS   (176 * 512)             // 90,112 halves
#define WLDS_BYTES  (W16_HALFS * 2)         // 180,224 bytes

// ---- LDS layout ----
#define BIAS_FLOATS 640
#define BIAS_BYTES  (BIAS_FLOATS * 4)                  // 2,560
#define STAGE_BYTES ((16*160 + 16*128 + 16*128) * 2)   // 13,312
#define SMEM_BYTES  (WLDS_BYTES + BIAS_BYTES + 8 * STAGE_BYTES)  // 289,280

// Hardware tanh (gfx1250 TRANS op v_tanh_f32) with safe fallback
__device__ inline float htanh(float x) {
#if __has_builtin(__builtin_amdgcn_tanhf)
  return __builtin_amdgcn_tanhf(x);
#else
  return tanhf(x);
#endif
}

// nan_to_num + clamp into f16-representable range
__device__ inline _Float16 scrub(float v) {
  v = (v != v) ? 0.0f : v;
  v = fminf(fmaxf(v, -65504.0f), 65504.0f);
  return (_Float16)v;
}

// Build a v16h WMMA fragment from two 16-byte LDS loads
__device__ inline v16h ld_frag(const _Float16* p0, const _Float16* p1) {
  v8h lo = *(const v8h*)p0;
  v8h hi = *(const v8h*)p1;
  return __builtin_shufflevector(lo, hi, 0,1,2,3,4,5,6,7,8,9,10,11,12,13,14,15);
}

// ---------------------------------------------------------------------------
// dense layer: [16 x K] (LDS f16) x [K x NT*16] (swizzled LDS f16), bias added
// in the epilogue (C starts at inline 0), tanh, write f16 activation to LDS.
// B fragments double-buffered so ds_load(B(nt+1)) precedes the WMMA on B(nt).
// ---------------------------------------------------------------------------
template<int KC, int NT>
__device__ inline void dense_act(const _Float16* in, int inStride,
                                 const _Float16* w, const float* biasLds,
                                 _Float16* out, int outStride, int lane)
{
  const int col = lane & 15;
  const int hi  = lane >> 4;
  // prefetch bias into registers (consumed only in the epilogue)
  float bv[NT];
#pragma unroll
  for (int nt = 0; nt < NT; ++nt) bv[nt] = biasLds[nt * 16 + col];

  v8f acc[NT];
#pragma unroll
  for (int nt = 0; nt < NT; ++nt) acc[nt] = (v8f)(0.0f);

#pragma unroll
  for (int kc = 0; kc < KC; ++kc) {
    const _Float16* pa = in + col * inStride + kc * 32 + hi * 8;
    v16h afrag = ld_frag(pa, pa + 16);
    const _Float16* pb = w + kc * NT * 512 + lane * 16;
    v16h bcur = ld_frag(pb, pb + 8);
#pragma unroll
    for (int nt = 0; nt < NT; ++nt) {
      v16h bnext = bcur;
      if (nt + 1 < NT) {
        const _Float16* pn = pb + (nt + 1) * 512;
        bnext = ld_frag(pn, pn + 8);
      }
      acc[nt] = __builtin_amdgcn_wmma_f32_16x16x32_f16(
          false, afrag, false, bcur, (short)0, acc[nt], false, false);
      bcur = bnext;
    }
  }
  const int row0 = hi * 8;
#pragma unroll
  for (int nt = 0; nt < NT; ++nt)
#pragma unroll
    for (int r = 0; r < 8; ++r)
      out[(row0 + r) * outStride + nt * 16 + col] = (_Float16)htanh(acc[nt][r] + bv[nt]);
}

// ---------------------------------------------------------------------------
// final layer: [16 x 128] x [128 x 64], +b2 in epilogue, tanh, scatter-add
// ---------------------------------------------------------------------------
__device__ inline void dense_scatter(const _Float16* in, const _Float16* w,
                                     const float* biasLds, const int* addrs,
                                     int tileBase, float* delta, int lane)
{
  const int col = lane & 15;
  const int hi  = lane >> 4;
  float bv[4];
#pragma unroll
  for (int nt = 0; nt < 4; ++nt) bv[nt] = biasLds[nt * 16 + col];

  v8f acc[4];
#pragma unroll
  for (int nt = 0; nt < 4; ++nt) acc[nt] = (v8f)(0.0f);

#pragma unroll
  for (int kc = 0; kc < 4; ++kc) {
    const _Float16* pa = in + col * 128 + kc * 32 + hi * 8;
    v16h afrag = ld_frag(pa, pa + 16);
    const _Float16* pb = w + kc * 4 * 512 + lane * 16;
    v16h bcur = ld_frag(pb, pb + 8);
#pragma unroll
    for (int nt = 0; nt < 4; ++nt) {
      v16h bnext = bcur;
      if (nt + 1 < 4) {
        const _Float16* pn = pb + (nt + 1) * 512;
        bnext = ld_frag(pn, pn + 8);
      }
      acc[nt] = __builtin_amdgcn_wmma_f32_16x16x32_f16(
          false, afrag, false, bcur, (short)0, acc[nt], false, false);
      bcur = bnext;
    }
  }
  const int row0 = hi * 8;
  int arow[8];
#pragma unroll
  for (int r = 0; r < 8; ++r) arow[r] = addrs[tileBase + row0 + r];
#pragma unroll
  for (int nt = 0; nt < 4; ++nt)
#pragma unroll
    for (int r = 0; r < 8; ++r) {
      float v = htanh(acc[nt][r] + bv[nt]);
      atomicAdd(delta + (size_t)arow[r] * D_DIM + nt * 16 + col, v);
    }
}

// ---------------------------------------------------------------------------
// Kernel 0: zero the delta accumulator (must run every launch/replay)
// ---------------------------------------------------------------------------
__global__ void zero_kernel(float4* p, int n4) {
  int i = blockIdx.x * blockDim.x + threadIdx.x;
  if (i < n4) p[i] = make_float4(0.f, 0.f, 0.f, 0.f);
}

// ---------------------------------------------------------------------------
// Kernel 1: convert fp32 weights to f16, pre-swizzled into B-fragment order.
// ---------------------------------------------------------------------------
__global__ void prep_kernel(const float* fW0, const float* fW1, const float* fW2,
                            const float* tW0, const float* tW1, const float* tW2,
                            _Float16* w16)
{
  int idx = blockIdx.x * blockDim.x + threadIdx.x;
  if (idx >= W16_HALFS) return;
  int b   = idx >> 9;          // block index 0..175
  int tid = idx & 511;
  int m   = b / 88;
  int rem = b % 88;
  const float* W; int Kact, Ncols, kc, nt;
  if (rem < 40)      { W = m ? tW0 : fW0; Kact = 153; Ncols = 128; kc = rem >> 3;        nt = rem & 7; }
  else if (rem < 72) { int r = rem - 40; W = m ? tW1 : fW1; Kact = 128; Ncols = 128; kc = r >> 3; nt = r & 7; }
  else               { int r = rem - 72; W = m ? tW2 : fW2; Kact = 128; Ncols = 64;  kc = r >> 2; nt = r & 3; }
  int lane = tid >> 4, j = tid & 15;
  int col = nt * 16 + (lane & 15);
  int k   = kc * 32 + (lane >> 4) * 8 + (j & 7) + (j >> 3) * 16;
  float v = (k < Kact) ? W[k * Ncols + col] : 0.0f;
  w16[idx] = (_Float16)v;
}

// ---------------------------------------------------------------------------
// Kernel 2: main edge kernel.
// ---------------------------------------------------------------------------
__global__ void __launch_bounds__(256)
edge_kernel(const int* __restrict__ addr_from, const int* __restrict__ addr_to,
            const float* __restrict__ h_local, const float* __restrict__ h_global,
            const float* __restrict__ x_local, const float* __restrict__ x_global,
            const float* __restrict__ tptr,
            const float* fb0, const float* fb1, const float* fb2,
            const float* tb0, const float* tb1, const float* tb2,
            const _Float16* __restrict__ w16, float* __restrict__ delta)
{
  extern __shared__ char smem[];
  _Float16* wlds = (_Float16*)smem;
  float*    blds = (float*)(smem + WLDS_BYTES);

  // cooperative copy of all swizzled f16 weights into LDS (L2-resident source)
  {
    const uint4* src = (const uint4*)w16;
    uint4* dst = (uint4*)smem;
    for (int i = threadIdx.x; i < WLDS_BYTES / 16; i += 256) dst[i] = src[i];
  }
  // biases into LDS: [m][320] = {b0[128], b1[128], b2[64]}
  for (int i = threadIdx.x; i < BIAS_FLOATS; i += 256) {
    int m = i / 320, r = i % 320;
    const float* b;
    int off;
    if (r < 128)      { b = m ? tb0 : fb0; off = r; }
    else if (r < 256) { b = m ? tb1 : fb1; off = r - 128; }
    else              { b = m ? tb2 : fb2; off = r - 256; }
    blds[i] = b[off];
  }
  __syncthreads();

  const int wave = threadIdx.x >> 5;
  const int lane = threadIdx.x & 31;
  _Float16* inBuf = (_Float16*)(smem + WLDS_BYTES + BIAS_BYTES + wave * STAGE_BYTES);
  _Float16* actA  = inBuf + 16 * 160;
  _Float16* actB  = actA + 16 * 128;
  const float tval = tptr[0];

  for (int tile = blockIdx.x * 8 + wave; tile < NTILES; tile += gridDim.x * 8) {
    // ---- stage the [16 x 160] f16 input tile into LDS ----
    for (int e = 0; e < 16; ++e) {
      int eg = tile * 16 + e;
      int af = addr_from[eg];
      int at = addr_to[eg];
      const float* hf = h_local + (size_t)af * D_DIM;
      const float* ht = h_local + (size_t)at * D_DIM;
      _Float16* row = inBuf + e * 160;
      row[lane]       = scrub(hf[lane]);
      row[32 + lane]  = scrub(hf[32 + lane]);
      row[64 + lane]  = scrub(ht[lane]);
      row[96 + lane]  = scrub(ht[32 + lane]);
      float vt;
      if (lane < 4)       vt = x_local[(size_t)eg * 4 + lane];
      else if (lane < 20) vt = h_global[lane - 4];
      else if (lane < 24) vt = x_global[lane - 20];
      else if (lane == 24) vt = tval;
      else                vt = 0.0f;
      row[128 + lane] = scrub(vt);
    }

    // ---- two MLPs (from / to) over the shared input tile ----
    for (int m = 0; m < 2; ++m) {
      const _Float16* wm = wlds + m * (88 * 512);
      const float* bm = blds + m * 320;
      const int* aptr = m ? addr_to : addr_from;
      dense_act<5, 8>(inBuf, 160, wm,             bm,       actA, 128, lane);
      dense_act<4, 8>(actA, 128, wm + 40 * 512,   bm + 128, actB, 128, lane);
      dense_scatter  (actB,      wm + 72 * 512,   bm + 256, aptr, tile * 16, delta, lane);
    }
  }
}

// ---------------------------------------------------------------------------
// Kernel 3: out = tanh(delta)
// ---------------------------------------------------------------------------
__global__ void final_kernel(const float4* __restrict__ delta, float4* __restrict__ out, int n4) {
  int i = blockIdx.x * blockDim.x + threadIdx.x;
  if (i < n4) {
    float4 v = delta[i];
    v.x = htanh(v.x); v.y = htanh(v.y); v.z = htanh(v.z); v.w = htanh(v.w);
    out[i] = v;
  }
}

extern "C" void kernel_launch(void* const* d_in, const int* in_sizes, int n_in,
                              void* d_out, int out_size, void* d_ws, size_t ws_size,
                              hipStream_t stream) {
  const int*   addr_from = (const int*)d_in[0];
  const int*   addr_to   = (const int*)d_in[1];
  const float* h_local   = (const float*)d_in[2];
  const float* h_global  = (const float*)d_in[3];
  const float* x_local   = (const float*)d_in[4];
  const float* x_global  = (const float*)d_in[5];
  const float* t         = (const float*)d_in[6];
  const float* fW0 = (const float*)d_in[7];  const float* fb0 = (const float*)d_in[8];
  const float* fW1 = (const float*)d_in[9];  const float* fb1 = (const float*)d_in[10];
  const float* fW2 = (const float*)d_in[11]; const float* fb2 = (const float*)d_in[12];
  const float* tW0 = (const float*)d_in[13]; const float* tb0 = (const float*)d_in[14];
  const float* tW1 = (const float*)d_in[15]; const float* tb1 = (const float*)d_in[16];
  const float* tW2 = (const float*)d_in[17]; const float* tb2 = (const float*)d_in[18];

  float*     delta = (float*)d_ws;
  _Float16*  w16   = (_Float16*)((char*)d_ws + DELTA_BYTES);
  float*     out   = (float*)d_out;

  {
    int n4 = N_OBJ * D_DIM / 4;
    zero_kernel<<<(n4 + 255) / 256, 256, 0, stream>>>((float4*)delta, n4);
  }
  prep_kernel<<<(W16_HALFS + 255) / 256, 256, 0, stream>>>(fW0, fW1, fW2, tW0, tW1, tW2, w16);
  edge_kernel<<<2048, 256, SMEM_BYTES, stream>>>(
      addr_from, addr_to, h_local, h_global, x_local, x_global, t,
      fb0, fb1, fb2, tb0, tb1, tb2, w16, delta);
  {
    int n4 = N_OBJ * D_DIM / 4;
    final_kernel<<<(n4 + 255) / 256, 256, 0, stream>>>((const float4*)delta, (float4*)out, n4);
  }
}
